// RKT_77369540870790
// MI455X (gfx1250) — matrix operand order repaired and checked by reading, not compiled, not run on hardware
//
#include <hip/hip_runtime.h>
#include <hip/hip_bf16.h>

// ---------------------------------------------------------------------------
// Sizes (match reference)
// ---------------------------------------------------------------------------
#define Bq 32
#define Sq 512
#define EMB 256
#define NH 8
#define HD 32
#define NTOK (Bq * Sq)          // 16384
#define NEG_BIG_F (-1000000000.0f)
#define NEG_SMALL_F (-100000.0f)

#define SCP (Sq + 4)            // padded f32 score row (516: +4 banks/row)
#define AFP (Sq + 8)            // padded f16 attn row  (520: +4 banks/row)

typedef __attribute__((ext_vector_type(16))) _Float16 v16h;
typedef __attribute__((ext_vector_type(8)))  _Float16 v8h;
typedef __attribute__((ext_vector_type(8)))  float    v8f;
typedef __attribute__((ext_vector_type(4)))  int      v4i;

// Optional gfx1250 async global->LDS path (probe via __has_builtin so the
// file compiles either way; fallback = direct coalesced global reads).
#if defined(__has_builtin)
#if __has_builtin(__builtin_amdgcn_global_load_async_to_lds_b128) && \
    __has_builtin(__builtin_amdgcn_s_wait_asynccnt)
#define USE_ASYNC_LDS 1
#endif
#endif
#ifndef USE_ASYNC_LDS
#define USE_ASYNC_LDS 0
#endif

// ---------------------------------------------------------------------------
// WMMA fragment loaders (wave32; contiguous k = 16*(lane/16) + s, same order in A and B)
// ---------------------------------------------------------------------------

// A-matrix 16x32 f16, row-major memory with leading dim `ld`.
__device__ inline v16h wmma_load_a(const _Float16* base, int ld) {
  int lane = threadIdx.x & 31;
  int row  = lane & 15;
  int kb   = (lane >> 4) << 4;           // 0 or 16
  const _Float16* p = base + (size_t)row * ld;
  v16h a;
#pragma unroll
  for (int h = 0; h < 8; ++h) a[h] = p[kb + h];
#pragma unroll
  for (int h = 0; h < 8; ++h) a[8 + h] = p[kb + 8 + h];
  return a;
}

// B-matrix 32x16 f16 where memory holds B transposed: baseT[n*ld + k].
__device__ inline v16h wmma_load_b_T(const _Float16* baseT, int ld) {
  int lane = threadIdx.x & 31;
  int n    = lane & 15;
  int kb   = (lane >> 4) << 4;           // 0 or 16
  const _Float16* p = baseT + (size_t)n * ld + kb;
  v16h b;
#pragma unroll
  for (int h = 0; h < 16; ++h) b[h] = p[h];
  return b;
}

// ---------------------------------------------------------------------------
// Kernel 0: weight convert+transpose  W(rows x cols) f32 -> WT(cols x rows) f16
// ---------------------------------------------------------------------------
__global__ void RKT_wconv(const float* __restrict__ W, _Float16* __restrict__ WT,
                          int rows, int cols) {
  int i = blockIdx.x * blockDim.x + threadIdx.x;
  if (i >= rows * cols) return;
  int r = i / cols, c = i - r * cols;
  WT[(size_t)c * rows + r] = (_Float16)W[i];
}

// ---------------------------------------------------------------------------
// Kernel 1: embedding gather; build X = [e*lab, e*(1-lab)] (f16) and Qe (f16)
// ---------------------------------------------------------------------------
__global__ void RKT_gather(const int* __restrict__ item_inputs,
                           const int* __restrict__ labels,
                           const int* __restrict__ item_ids,
                           const float* __restrict__ embeds,
                           _Float16* __restrict__ X,
                           _Float16* __restrict__ Qe) {
  int row = blockIdx.x;            // token index 0..16383
  int t   = threadIdx.x;           // 0..255
  int id  = item_inputs[row];
  float lab = (float)labels[row];
  float e = embeds[(size_t)id * EMB + t];
  X[(size_t)row * (2 * EMB) + t]       = (_Float16)(e * lab);
  X[(size_t)row * (2 * EMB) + EMB + t] = (_Float16)(e * (1.0f - lab));
  int qid = item_ids[row];
  Qe[(size_t)row * EMB + t] = (_Float16)embeds[(size_t)qid * EMB + t];
}

// ---------------------------------------------------------------------------
// Kernel 2: WMMA GEMM  C = act(A(f16,MxK) @ B(via BT f16) + bias), f16 out.
// storeT=0: C row-major (MxN). storeT=1: store C transposed (NxM) with one
// packed 16B store per lane (contiguous 8 rows per column).
// ---------------------------------------------------------------------------
__global__ void RKT_gemm(const _Float16* __restrict__ A,
                         const _Float16* __restrict__ BT,
                         const float* __restrict__ bias,
                         _Float16* __restrict__ C,
                         int M, int N, int Kd, int relu, int storeT) {
  int wave = threadIdx.x >> 5;
  int lane = threadIdx.x & 31;
  int m0 = blockIdx.x * 16;
  int n0 = blockIdx.y * 64 + wave * 16;
  v8f c = {};
  for (int k = 0; k < Kd; k += 32) {
    v16h a = wmma_load_a(A + (size_t)m0 * Kd + k, Kd);
    v16h b = wmma_load_b_T(BT + (size_t)n0 * Kd + k, Kd);
    c = __builtin_amdgcn_wmma_f32_16x16x32_f16(false, a, false, b,
                                               (short)0, c, false, false);
  }
  int col = n0 + (lane & 15);
  int rb  = (lane >> 4) * 8;
  float bv = bias[col];
  if (storeT) {
    v8h pk;
#pragma unroll
    for (int v = 0; v < 8; ++v) {
      float val = c[v] + bv;
      if (relu) val = fmaxf(val, 0.0f);
      pk[v] = (_Float16)val;
    }
    *(v8h*)(C + (size_t)col * M + m0 + rb) = pk;   // 16B aligned
  } else {
#pragma unroll
    for (int v = 0; v < 8; ++v) {
      float val = c[v] + bv;
      if (relu) val = fmaxf(val, 0.0f);
      C[(size_t)(m0 + rb + v) * N + col] = (_Float16)val;
    }
  }
}

// ---------------------------------------------------------------------------
// Kernel 3: head-independent rel_attn and time_attn rows (computed once)
// ---------------------------------------------------------------------------
__global__ void RKT_relts(const float* __restrict__ rel,
                          const float* __restrict__ ts,
                          float* __restrict__ relA,
                          float* __restrict__ timA) {
  __shared__ float red[256];
  int row = blockIdx.x;            // b*S + q
  int q   = row & (Sq - 1);
  int t   = threadIdx.x;
  size_t base = (size_t)row * Sq;
  int c0 = t, c1 = t + 256;

  float v0 = (c0 <= q) ? rel[base + c0] : 0.0f;
  float v1 = (c1 <= q) ? rel[base + c1] : 0.0f;
  float x0 = (v0 == 0.0f) ? NEG_SMALL_F : v0;
  float x1 = (v1 == 0.0f) ? NEG_SMALL_F : v1;
  float y0 = (c0 > q) ? NEG_SMALL_F : __expf(-fabsf(ts[base + c0]));
  float y1 = (c1 > q) ? NEG_SMALL_F : __expf(-fabsf(ts[base + c1]));

  red[t] = fmaxf(x0, x1); __syncthreads();
  for (int o = 128; o > 0; o >>= 1) { if (t < o) red[t] = fmaxf(red[t], red[t + o]); __syncthreads(); }
  float m = red[0]; __syncthreads();
  float e0 = __expf(x0 - m), e1 = __expf(x1 - m);
  red[t] = e0 + e1; __syncthreads();
  for (int o = 128; o > 0; o >>= 1) { if (t < o) red[t] += red[t + o]; __syncthreads(); }
  float inv = 1.0f / red[0]; __syncthreads();
  relA[base + c0] = e0 * inv;
  relA[base + c1] = e1 * inv;

  red[t] = fmaxf(y0, y1); __syncthreads();
  for (int o = 128; o > 0; o >>= 1) { if (t < o) red[t] = fmaxf(red[t], red[t + o]); __syncthreads(); }
  m = red[0]; __syncthreads();
  e0 = __expf(y0 - m); e1 = __expf(y1 - m);
  red[t] = e0 + e1; __syncthreads();
  for (int o = 128; o > 0; o >>= 1) { if (t < o) red[t] += red[t + o]; __syncthreads(); }
  inv = 1.0f / red[0];
  timA[base + c0] = e0 * inv;
  timA[base + c1] = e1 * inv;
}

// ---------------------------------------------------------------------------
// async helper: copy one 512-float row global -> LDS (wave-cooperative, 4x16B
// per lane); tracked by ASYNCcnt.  Builtin prototype (from the compiler
// diagnostic): (v4i addrspace(1)*, v4i addrspace(3)*, imm int, imm int).
// Go through integers to materialize the address-space-qualified pointers:
// global generic address == addrspace(1) address; low 32 bits of a generic
// LDS address are the LDS offset on CDNA5.
// ---------------------------------------------------------------------------
#if USE_ASYNC_LDS
typedef __attribute__((address_space(1))) v4i* as1_v4i_ptr;
typedef __attribute__((address_space(3))) v4i* as3_v4i_ptr;

__device__ inline void stage_row_async(const float* g, float* l) {
  int lane = threadIdx.x & 31;
  const float* gs = g + lane * 4;
  float* ls = l + lane * 4;
#pragma unroll
  for (int ch = 0; ch < 4; ++ch) {
    __builtin_amdgcn_global_load_async_to_lds_b128(
        (as1_v4i_ptr)(size_t)(gs + ch * 128),
        (as3_v4i_ptr)(unsigned)(size_t)(ls + ch * 128),
        0, 0);
  }
}
#endif

// ---------------------------------------------------------------------------
// Kernel 4: fused attention. 1 wave per (b,h,16-row q-tile).
// ---------------------------------------------------------------------------
__global__ void RKT_attn(const _Float16* __restrict__ Q,
                         const _Float16* __restrict__ K,
                         const _Float16* __restrict__ VT,   // (EMB x NTOK) f16
                         const float* __restrict__ relA,
                         const float* __restrict__ timA,
                         const float* __restrict__ l1p,
                         const float* __restrict__ l2p,
                         float* __restrict__ attn_out,
                         float* __restrict__ OUT) {
  __shared__ float    sc[16][SCP];     // padded score strip (~33 KB)
  __shared__ _Float16 af[16][AFP];     // padded f16 attn (~16.6 KB)
  __shared__ float    smax[16], sinv[16];
#if USE_ASYNC_LDS
  __shared__ float relbuf[2][Sq];      // double-buffered async staging (8 KB)
  __shared__ float timbuf[2][Sq];
#endif

  int qt = blockIdx.x & 31;
  int h  = (blockIdx.x >> 5) & 7;
  int b  = blockIdx.x >> 8;
  int lane = threadIdx.x & 31;
  int rb = (lane >> 4) * 8;
  int cl = lane & 15;
  const float scale = 0.17677669529663689f; // 1/sqrt(32)

  // --- scores: Q(16x32) . K^T, 32 key tiles ---
  const _Float16* Qb = Q + ((size_t)(b * Sq + qt * 16) * EMB) + h * HD;
  v16h aq = wmma_load_a(Qb, EMB);
  for (int kt = 0; kt < 32; ++kt) {
    const _Float16* Kb = K + ((size_t)(b * Sq + kt * 16) * EMB) + h * HD;
    v16h bf = wmma_load_b_T(Kb, EMB);
    v8f c = {};
    c = __builtin_amdgcn_wmma_f32_16x16x32_f16(false, aq, false, bf,
                                               (short)0, c, false, false);
    int key = kt * 16 + cl;
#pragma unroll
    for (int v = 0; v < 8; ++v) {
      int qg = qt * 16 + rb + v;
      float val = c[v] * scale;
      if (key > qg) val = NEG_BIG_F;
      sc[rb + v][key] = val;
    }
  }
  __syncthreads();

  // --- softmax stats: 2 lanes per row, combine via shfl_xor(16) ---
  {
    int r  = lane & 15;
    int cb = (lane >> 4) * 256;
    float m = -3.4e38f;
    for (int j = 0; j < 256; ++j) m = fmaxf(m, sc[r][cb + j]);
    m = fmaxf(m, __shfl_xor(m, 16, 32));
    float s = 0.0f;
    for (int j = 0; j < 256; ++j) s += __expf(sc[r][cb + j] - m);
    s += __shfl_xor(s, 16, 32);
    if (lane < 16) { smax[lane] = m; sinv[lane] = 1.0f / s; }
  }
  __syncthreads();

  float l1 = l1p[0], l2 = l2p[0];
  float w_prob = (1.0f - l1) * (1.0f - l2);
  float w_time = (1.0f - l1) * l2;

#if USE_ASYNC_LDS
  {
    size_t rb0 = (size_t)(b * Sq + qt * 16) * Sq;
    stage_row_async(relA + rb0, &relbuf[0][0]);
    stage_row_async(timA + rb0, &timbuf[0][0]);
  }
#endif

  // --- blend + write attn: whole wave sweeps one row -> coalesced 128B ops ---
  for (int r = 0; r < 16; ++r) {
    int qg = qt * 16 + r;
    float m = smax[r], inv = sinv[r];
    float* ao = attn_out + (((size_t)(b * NH + h) * Sq + qg) * Sq);
#if USE_ASYNC_LDS
    if (r < 15) {
      size_t rbn = (size_t)(b * Sq + qg + 1) * Sq;
      stage_row_async(relA + rbn, &relbuf[(r + 1) & 1][0]);
      stage_row_async(timA + rbn, &timbuf[(r + 1) & 1][0]);
      __builtin_amdgcn_s_wait_asynccnt(8);   // current row's 8 ops done
    } else {
      __builtin_amdgcn_s_wait_asynccnt(0);
    }
    const float* rrow = &relbuf[r & 1][0];
    const float* trow = &timbuf[r & 1][0];
#else
    const float* rrow = relA + (size_t)(b * Sq + qg) * Sq;
    const float* trow = timA + (size_t)(b * Sq + qg) * Sq;
#endif
    for (int c = lane; c < Sq; c += 32) {
      float prob = __expf(sc[r][c] - m) * inv;
      float blend = w_prob * prob + w_time * trow[c] + l1 * rrow[c];
      ao[c] = blend;                      // coalesced 256MB output write
      af[r][c] = (_Float16)blend;
    }
  }
  __syncthreads();

  // --- out tile = attn(16x512) @ V(512x32), B from transposed V (contig) ---
  v8f c0 = {}, c1 = {};
  for (int kt = 0; kt < 16; ++kt) {
    v16h aa = wmma_load_a(&af[0][kt * 32], AFP);
    const _Float16* Vb0 = VT + (size_t)(h * HD) * NTOK + (b * Sq + kt * 32);
    const _Float16* Vb1 = VT + (size_t)(h * HD + 16) * NTOK + (b * Sq + kt * 32);
    v16h b0 = wmma_load_b_T(Vb0, NTOK);
    v16h b1 = wmma_load_b_T(Vb1, NTOK);
    c0 = __builtin_amdgcn_wmma_f32_16x16x32_f16(false, aa, false, b0,
                                                (short)0, c0, false, false);
    c1 = __builtin_amdgcn_wmma_f32_16x16x32_f16(false, aa, false, b1,
                                                (short)0, c1, false, false);
  }
#pragma unroll
  for (int v = 0; v < 8; ++v) {
    int qq = qt * 16 + rb + v;
    float* ob = OUT + ((size_t)(b * Sq + qq) * EMB) + h * HD;
    ob[cl]      = c0[v];
    ob[16 + cl] = c1[v];
  }
}

// ---------------------------------------------------------------------------
// Kernel 5: logits = OUT @ W_out + b_out
// ---------------------------------------------------------------------------
__global__ void RKT_logits(const float* __restrict__ OUT,
                           const float* __restrict__ Wout,
                           const float* __restrict__ bout,
                           float* __restrict__ logits) {
  __shared__ float red[256];
  int row = blockIdx.x;
  int t = threadIdx.x;
  red[t] = OUT[(size_t)row * EMB + t] * Wout[t];
  __syncthreads();
  for (int o = 128; o > 0; o >>= 1) { if (t < o) red[t] += red[t + o]; __syncthreads(); }
  if (t == 0) logits[row] = red[0] + bout[0];
}

// ---------------------------------------------------------------------------
// Host launcher
// ---------------------------------------------------------------------------
extern "C" void kernel_launch(void* const* d_in, const int* in_sizes, int n_in,
                              void* d_out, int out_size, void* d_ws, size_t ws_size,
                              hipStream_t stream) {
  const int*   item_inputs = (const int*)d_in[0];
  const int*   labels      = (const int*)d_in[1];
  const int*   item_ids    = (const int*)d_in[2];
  const float* rel         = (const float*)d_in[3];
  const float* ts          = (const float*)d_in[4];
  const float* embeds      = (const float*)d_in[5];
  const float* W_in        = (const float*)d_in[6];
  const float* b_in        = (const float*)d_in[7];
  const float* Wq          = (const float*)d_in[8];
  const float* bq          = (const float*)d_in[9];
  const float* Wk          = (const float*)d_in[10];
  const float* bk          = (const float*)d_in[11];
  const float* Wv          = (const float*)d_in[12];
  const float* bv          = (const float*)d_in[13];
  const float* W_out       = (const float*)d_in[14];
  const float* b_out       = (const float*)d_in[15];
  const float* l1p         = (const float*)d_in[16];
  const float* l2p         = (const float*)d_in[17];

  float* out    = (float*)d_out;
  float* logits = out;                        // 16384 floats
  float* attn   = out + (size_t)NTOK;         // 67108864 floats

  char* ws = (char*)d_ws;
  auto carve = [&](size_t bytes) {
    char* p = ws; ws += (bytes + 255) & ~(size_t)255; return p;
  };
  _Float16* X    = (_Float16*)carve((size_t)NTOK * 2 * EMB * 2);   // 16 MB
  _Float16* Qe   = (_Float16*)carve((size_t)NTOK * EMB * 2);       //  8 MB
  _Float16* Hm   = (_Float16*)carve((size_t)NTOK * EMB * 2);       //  8 MB
  _Float16* Qm   = (_Float16*)carve((size_t)NTOK * EMB * 2);       //  8 MB
  _Float16* Km   = (_Float16*)carve((size_t)NTOK * EMB * 2);       //  8 MB
  _Float16* VT   = (_Float16*)carve((size_t)NTOK * EMB * 2);       //  8 MB (EMBxNTOK)
  _Float16* WinT = (_Float16*)carve((size_t)2 * EMB * EMB * 2);
  _Float16* WqT  = (_Float16*)carve((size_t)EMB * EMB * 2);
  _Float16* WkT  = (_Float16*)carve((size_t)EMB * EMB * 2);
  _Float16* WvT  = (_Float16*)carve((size_t)EMB * EMB * 2);
  float*    relA = (float*)carve((size_t)NTOK * Sq * 4);           // 32 MB
  float*    timA = (float*)carve((size_t)NTOK * Sq * 4);           // 32 MB
  float*    OUT  = (float*)carve((size_t)NTOK * EMB * 4);          // 16 MB

  RKT_wconv<<<(2 * EMB * EMB + 255) / 256, 256, 0, stream>>>(W_in, WinT, 2 * EMB, EMB);
  RKT_wconv<<<(EMB * EMB + 255) / 256, 256, 0, stream>>>(Wq, WqT, EMB, EMB);
  RKT_wconv<<<(EMB * EMB + 255) / 256, 256, 0, stream>>>(Wk, WkT, EMB, EMB);
  RKT_wconv<<<(EMB * EMB + 255) / 256, 256, 0, stream>>>(Wv, WvT, EMB, EMB);

  RKT_gather<<<NTOK, 256, 0, stream>>>(item_inputs, labels, item_ids, embeds, X, Qe);

  dim3 ggrid(NTOK / 16, EMB / 64);
  RKT_gemm<<<ggrid, 128, 0, stream>>>(X,  WinT, b_in, Hm, NTOK, EMB, 2 * EMB, 1, 0);
  RKT_gemm<<<ggrid, 128, 0, stream>>>(Qe, WqT,  bq,   Qm, NTOK, EMB, EMB,     0, 0);
  RKT_gemm<<<ggrid, 128, 0, stream>>>(Hm, WkT,  bk,   Km, NTOK, EMB, EMB,     0, 0);
  RKT_gemm<<<ggrid, 128, 0, stream>>>(Hm, WvT,  bv,   VT, NTOK, EMB, EMB,     0, 1);

  RKT_relts<<<NTOK, 256, 0, stream>>>(rel, ts, relA, timA);

  RKT_attn<<<Bq * NH * (Sq / 16), 32, 0, stream>>>(Qm, Km, VT, relA, timA,
                                                   l1p, l2p, attn, OUT);

  RKT_logits<<<NTOK, 256, 0, stream>>>(OUT, W_out, b_out, logits);
}
